// CustomNet_59390807769139
// MI455X (gfx1250) — matrix operand order repaired and checked by reading, not compiled, not run on hardware
//
#include <hip/hip_runtime.h>

// ---------------------------------------------------------------------------
// Sparse 3D conv (16->32 masked ->16) for MI455X / gfx1250, wave32 + WMMA.
//
//  * dense / h / occ grids carry a 1-voxel zero halo (34 x 130 x 130) so SAME
//    padding needs NO bounds checks: every conv tap is base + constant offset
//    folded into the global-load IOFFSET (branchless, EXEC always all-ones).
//  * conv1 is dense (dilated mask ~93% occupancy), conv2 is driven by the
//    200k voxel list (10x less compute than dense conv2 would be).
//  * all matrix math via v_wmma_f32_16x16x32_f16 (f16 in, f32 accumulate).
//  * WMMA C fragments are staged through LDS (ds_store_b16 swizzle ->
//    s_wait_dscnt -> b128 coalesced global stores).
//  * intermediates (~221MB) mostly live in the 192MB L2; HBM floor ~25us.
// ---------------------------------------------------------------------------

typedef _Float16 h16 __attribute__((ext_vector_type(16)));
typedef _Float16 h8  __attribute__((ext_vector_type(8)));
typedef float    f8  __attribute__((ext_vector_type(8)));
typedef float    f4  __attribute__((ext_vector_type(4)));
typedef unsigned int u4v __attribute__((ext_vector_type(4)));

#define BB   4
#define DDZ  32
#define HHY  128
#define WWX  128
#define CI   16
#define CO   32
#define NVOX 200000

// padded grid (1-voxel halo each side)
#define PD   (DDZ + 2)            // 34
#define PH   (HHY + 2)            // 130
#define PW   (WWX + 2)            // 130
#define PPL  (PH * PW)            // 16900
#define PBT  (PD * PPL)           // 574600
#define NPV  (BB * PBT)           // padded voxels total

union A16 { h16 v; h8 h[2]; u4v u[2]; };

// linear padded index of UNPADDED coords (b, z, y, x)
static __device__ __forceinline__ long vox_pad(int b, int z, int y, int x) {
  return (long)b * PBT + (long)(z + 1) * PPL + (long)(y + 1) * PW + (x + 1);
}
// constant tap offset (in padded voxels) for tap t in [0,27)
static __device__ __forceinline__ constexpr int tap_off(int t) {
  return (t / 9 - 1) * PPL + ((t / 3) % 3 - 1) * PW + (t % 3 - 1);
}

// ---------------------------------------------------------------- zero init
__global__ void k_zero(u4v* __restrict__ dense4, u4v* __restrict__ occ4,
                       int nd4, int no4) {
  int i = blockIdx.x * blockDim.x + threadIdx.x;
  const u4v z = {0u, 0u, 0u, 0u};
  if (i < nd4) {
    dense4[i] = z;
  } else {
    int j = i - nd4;
    if (j < no4) occ4[j] = z;
  }
}

// zero only the halo voxels of h (interior is fully overwritten by conv1)
__global__ void k_halo(_Float16* __restrict__ hbuf) {
  int i = blockIdx.x * blockDim.x + threadIdx.x;
  if (i >= NPV) return;
  int x = i % PW; int t = i / PW;
  int y = t % PH; t /= PH;
  int z = t % PD;
  if (z == 0 || z == PD - 1 || y == 0 || y == PH - 1 || x == 0 || x == PW - 1) {
    u4v* p = (u4v*)(hbuf + (long)i * CO);
    const u4v zz = {0u, 0u, 0u, 0u};
    p[0] = zz; p[1] = zz; p[2] = zz; p[3] = zz;
  }
}

// ------------------------------------------- weight f16 convert + B-swizzle
// B fragment (16-bit 32x16): lanes 0-15 hold K=0..7,16..23 (elem h<8 <-> K=h, else K=h+8),
// lanes 16-31 hold K=8..15,24..31; stored per-lane-contiguous (one 32B h16 load).
__global__ void k_pack(const float* __restrict__ W1, const float* __restrict__ W2,
                       _Float16* __restrict__ w1b, _Float16* __restrict__ w2b) {
  int idx = blockIdx.x * blockDim.x + threadIdx.x;
  if (idx < 14336) {                 // W1: 14 tap-pairs x 2 N-tiles x 32 lanes x 16
    int h    = idx & 15;
    int lane = (idx >> 4) & 31;
    int s    = idx >> 9;             // s = kk*2 + nt
    int nt   = s & 1;
    int kk   = s >> 1;
    int g    = lane >> 4;
    int n    = lane & 15;
    int k    = (h < 8) ? (8 * g + h) : (16 + 8 * g + (h - 8));
    int tap  = 2 * kk + (k >> 4);
    int ci   = k & 15;
    int co   = nt * 16 + n;
    float v  = (tap < 27) ? W1[((long)tap * CI + ci) * CO + co] : 0.f;
    w1b[idx] = (_Float16)v;
  } else {
    int j = idx - 14336;             // W2: 27 taps x 32 lanes x 16
    if (j < 13824) {
      int h    = j & 15;
      int lane = (j >> 4) & 31;
      int t    = j >> 9;
      int g    = lane >> 4;
      int n    = lane & 15;
      int ci   = (h < 8) ? (8 * g + h) : (16 + 8 * g + (h - 8));
      w2b[j]   = (_Float16)W2[((long)t * CO + ci) * CI + n];
    }
  }
}

// --------------------------------------------------------- scatter f32->f16
// reference coords come from a permutation => unique => stores, not atomics.
__global__ void k_scatter(const float* __restrict__ feats, const int* __restrict__ coors,
                          _Float16* __restrict__ dense, unsigned char* __restrict__ occ) {
  int i = blockIdx.x * blockDim.x + threadIdx.x;
  if (i >= NVOX) return;
  int b = coors[i * 4 + 0], z = coors[i * 4 + 1];
  int y = coors[i * 4 + 2], x = coors[i * 4 + 3];
  long v = vox_pad(b, z, y, x);
  h8 lo, hi;
#pragma unroll
  for (int c = 0; c < 8; ++c) {
    lo[c] = (_Float16)feats[i * CI + c];
    hi[c] = (_Float16)feats[i * CI + 8 + c];
  }
  *(h8*)(dense + v * CI)     = lo;
  *(h8*)(dense + v * CI + 8) = hi;
  occ[v] = 1;
}

// ------------------------------------------------------- conv1: dense 16->32
// One wave per 16-voxel x-strip.  M=16 voxels, N=16 (x2 for CO=32),
// K=32 = 2 taps x 16 channels, 14 K-steps (tap 27 zero padded).
// Branchless: all tap loads are base + constant IOFFSET into padded grids.
__global__ void __launch_bounds__(256)
k_conv1(const _Float16* __restrict__ dense, const _Float16* __restrict__ w1b,
        const unsigned char* __restrict__ occ, _Float16* __restrict__ hbuf) {
  __shared__ _Float16 stage[8][16 * CO];     // 1KB per wave
  const int lane = threadIdx.x & 31;
  const int wib  = threadIdx.x >> 5;
  int t = blockIdx.x * 8 + wib;
  const int xt = t & 7;   t >>= 3;           // WWX/16 = 8
  const int y  = t & 127; t >>= 7;           // HHY = 128
  const int z  = t & 31;  t >>= 5;           // DDZ = 32
  const int b  = t;
  const int g  = lane >> 4;
  const int m  = lane & 15;
  const int x  = (xt << 4) + m;

  const long vc = vox_pad(b, z, y, x);

  // dilated occupancy (reference mask_out), branchless on padded occ
  const unsigned char* ob = occ + vc;
  int any = 0;
#pragma unroll
  for (int tt = 0; tt < 27; ++tt) any |= ob[tap_off(tt)];
  const unsigned int rowmask = (unsigned int)__ballot(any != 0) & 0xFFFFu;

  const _Float16* ab = dense + vc * CI + 8 * g;  // A: lane's channel half
  const _Float16* bb = w1b + lane * 16;          // B: pre-swizzled per lane

  f8 acc0 = {};
  f8 acc1 = {};
  const u4v z4 = {0u, 0u, 0u, 0u};
#pragma unroll
  for (int kk = 0; kk < 14; ++kk) {
    A16 a;
    a.h[0] = *(const h8*)(ab + (long)tap_off(2 * kk) * CI);
    if (2 * kk + 1 < 27)
      a.h[1] = *(const h8*)(ab + (long)tap_off(2 * kk + 1) * CI);
    else
      a.u[1] = z4;
    const h16 b0 = *(const h16*)(bb + (2 * kk)     * 512);
    const h16 b1 = *(const h16*)(bb + (2 * kk + 1) * 512);
    acc0 = __builtin_amdgcn_wmma_f32_16x16x32_f16(false, a.v, false, b0,
                                                  (short)0, acc0, false, false);
    acc1 = __builtin_amdgcn_wmma_f32_16x16x32_f16(false, a.v, false, b1,
                                                  (short)0, acc1, false, false);
  }

  // stage C (voxel-major) in LDS, then coalesced b128 stores: the strip's
  // 16 voxels x 32ch are contiguous in the padded h layout.
  _Float16* slot = &stage[wib][0];
#pragma unroll
  for (int r = 0; r < 8; ++r) {
    const int mm = r + 8 * g;                 // C: VGPR r -> row r + 8*(lane/16)
    const bool keep = ((rowmask >> mm) & 1u) != 0u;
    slot[mm * CO + m]      = keep ? (_Float16)acc0[r] : (_Float16)0.f;
    slot[mm * CO + 16 + m] = keep ? (_Float16)acc1[r] : (_Float16)0.f;
  }
  asm volatile("s_wait_dscnt 0x0" ::: "memory");

  _Float16* dst = hbuf + (vc - m) * CO;       // voxel x0 base
  const h8* s8 = (const h8*)slot;             // 64 x 16B chunks
  *(h8*)(dst + lane * 8)        = s8[lane];
  *(h8*)(dst + (lane + 32) * 8) = s8[lane + 32];
}

// ----------------------------------------------- conv2: list-driven 32->16
// One wave per 16 listed voxels; K=32 = one tap's channels => 27 WMMAs.
__global__ void __launch_bounds__(256)
k_conv2(const _Float16* __restrict__ hbuf, const _Float16* __restrict__ w2b,
        const int* __restrict__ coors, float* __restrict__ out) {
  __shared__ float stage[8][256];             // 1KB per wave
  const int lane = threadIdx.x & 31;
  const int wib  = threadIdx.x >> 5;
  const int tile = blockIdx.x * 8 + wib;
  if (tile >= NVOX / 16) return;              // wave-uniform exit
  const int g  = lane >> 4;
  const int m  = lane & 15;
  const int vi = tile * 16 + m;
  const int b = coors[vi * 4 + 0], z = coors[vi * 4 + 1];
  const int y = coors[vi * 4 + 2], x = coors[vi * 4 + 3];

  const long vc = vox_pad(b, z, y, x);
  const _Float16* ab = hbuf + vc * CO + 8 * g;
  const _Float16* bb = w2b + lane * 16;

  f8 acc = {};
#pragma unroll
  for (int t = 0; t < 27; ++t) {
    A16 a;
    a.h[0] = *(const h8*)(ab + (long)tap_off(t) * CO);        // K 0..7 / 8..15
    a.h[1] = *(const h8*)(ab + (long)tap_off(t) * CO + 16);   // K 16..23 / 24..31
    const h16 bm = *(const h16*)(bb + t * 512);
    acc = __builtin_amdgcn_wmma_f32_16x16x32_f16(false, a.v, false, bm,
                                                 (short)0, acc, false, false);
  }

  float* slot = &stage[wib][0];
#pragma unroll
  for (int r = 0; r < 8; ++r)
    slot[(r + 8 * g) * 16 + m] = acc[r];
  asm volatile("s_wait_dscnt 0x0" ::: "memory");

  float* dst = out + (long)tile * 256;        // 16 voxels x 16ch, contiguous
  ((f4*)dst)[lane]      = ((const f4*)slot)[lane];
  ((f4*)dst)[lane + 32] = ((const f4*)slot)[lane + 32];
}

// ---------------------------------------------------------------------------
extern "C" void kernel_launch(void* const* d_in, const int* in_sizes, int n_in,
                              void* d_out, int out_size, void* d_ws, size_t ws_size,
                              hipStream_t stream) {
  (void)in_sizes; (void)n_in; (void)out_size; (void)ws_size;
  const float* feats = (const float*)d_in[0];
  const int*   coors = (const int*)d_in[1];
  const float* W1    = (const float*)d_in[2];
  const float* W2    = (const float*)d_in[3];
  float* out = (float*)d_out;
  char*  ws  = (char*)d_ws;

  const size_t nDense = (size_t)NPV * CI;      // padded f16 grid (73.5MB)
  const size_t nH     = (size_t)NPV * CO;      // padded f16 grid (147MB)
  const size_t nOcc   = (size_t)NPV;           // padded occupancy (2.3MB)
  size_t off = 0;
  _Float16*      dense = (_Float16*)(ws + off);      off += nDense * 2;
  _Float16*      hbuf  = (_Float16*)(ws + off);      off += nH * 2;
  unsigned char* occ   = (unsigned char*)(ws + off); off += nOcc;
  _Float16*      w1b   = (_Float16*)(ws + off);      off += 14336 * 2;
  _Float16*      w2b   = (_Float16*)(ws + off);      off += 13824 * 2;
  // total ~223MB of d_ws

  const int nd4 = (int)(nDense * 2 / 16);
  const int no4 = (int)(nOcc / 16);
  const int nz  = nd4 + no4;
  k_zero<<<(nz + 255) / 256, 256, 0, stream>>>((u4v*)dense, (u4v*)occ, nd4, no4);
  k_halo<<<(NPV + 255) / 256, 256, 0, stream>>>(hbuf);
  k_pack<<<110, 256, 0, stream>>>(W1, W2, w1b, w2b);
  k_scatter<<<(NVOX + 255) / 256, 256, 0, stream>>>(feats, coors, dense, occ);

  const int strips = BB * DDZ * HHY * (WWX / 16);   // 131072 waves
  k_conv1<<<strips / 8, 256, 0, stream>>>(dense, w1b, occ, hbuf);

  const int tiles = NVOX / 16;                      // 12500 waves
  k_conv2<<<(tiles + 7) / 8, 256, 0, stream>>>(hbuf, w2b, coors, out);
}
